// GenerativeNet_style_9491877724181
// MI455X (gfx1250) — matrix operand order repaired and checked by reading, not compiled, not run on hardware
//
#include <hip/hip_runtime.h>
#include <hip/hip_bf16.h>
#include <math.h>

// ---------------------------------------------------------------------------
// Problem constants (match reference)
// ---------------------------------------------------------------------------
#define X_DIM 30000
#define Z_DIM 256
#define Y_DIM 64
#define B_DIM 4096
#define BN_EPS 1e-3f

typedef __attribute__((ext_vector_type(16))) __bf16 v16bf;
typedef __attribute__((ext_vector_type(8)))  float  v8f;

// ---------------------------------------------------------------------------
// Helpers
// ---------------------------------------------------------------------------
static __device__ __forceinline__ void split_bf16(float x, __bf16& hi, __bf16& lo) {
    hi = (__bf16)x;
    lo = (__bf16)(x - (float)hi);
}

// Load 16 bf16 as two contiguous 8-element (16B) chunks.
static __device__ __forceinline__ v16bf ld16(const __bf16* p0, const __bf16* p1) {
    union { uint4 u; __bf16 h[8]; } c0, c1;
    c0.u = *reinterpret_cast<const uint4*>(p0);
    c1.u = *reinterpret_cast<const uint4*>(p1);
    v16bf r;
#pragma unroll
    for (int i = 0; i < 8; ++i) { r[i] = c0.h[i]; r[8 + i] = c1.h[i]; }
    return r;
}

static __device__ __forceinline__ v8f wmma_bf16(v16bf a, v16bf b, v8f c) {
    // (neg_a, A, neg_b, B, c_mod, C, reuse_a, reuse_b)
    return __builtin_amdgcn_wmma_f32_16x16x32_bf16(false, a, false, b, (short)0, c,
                                                   false, false);
}

// ---------------------------------------------------------------------------
// Kernel 1: W_g2 (fp32, [X,Z] row-major) -> bf16 hi/lo split (same layout)
// ---------------------------------------------------------------------------
__global__ void k_conv_w(const float* __restrict__ W,
                         __bf16* __restrict__ hi, __bf16* __restrict__ lo, int n) {
    int i = blockIdx.x * blockDim.x + threadIdx.x;
    if (i < n) {
        float w = W[i];
        __bf16 h, l;
        split_bf16(w, h, l);
        hi[i] = h; lo[i] = l;
    }
}

// ---------------------------------------------------------------------------
// Kernel 2: transpose style_R [Y,X] -> Rt [X,Y] bf16 hi/lo; zero column stats
// ---------------------------------------------------------------------------
__global__ void k_prep_R(const float* __restrict__ R,
                         __bf16* __restrict__ Rhi, __bf16* __restrict__ Rlo,
                         float* __restrict__ s1, float* __restrict__ s2) {
    int x = blockIdx.x * blockDim.x + threadIdx.x;
    if (x >= X_DIM) return;
    s1[x] = 0.0f;
    s2[x] = 0.0f;
#pragma unroll
    for (int j = 0; j < Y_DIM; ++j) {
        float v = R[(size_t)j * X_DIM + x];
        __bf16 h, l;
        split_bf16(v, h, l);
        Rhi[(size_t)x * Y_DIM + j] = h;
        Rlo[(size_t)x * Y_DIM + j] = l;
    }
}

// ---------------------------------------------------------------------------
// Kernel 3: per batch-row front end.
//   y_mu = y@W_mu.T + b_mu ; y_var = softplus(y@W_var.T + b_var)
//   zs = softmax(z) ; tmp_L = y@style_L ; A = zs*tmp_L -> bf16 hi/lo
//   y -> bf16 hi/lo
// Grid: B blocks of Z threads.
// ---------------------------------------------------------------------------
__global__ void __launch_bounds__(Z_DIM)
k_front(const float* __restrict__ z, const float* __restrict__ y,
        const float* __restrict__ W_mu, const float* __restrict__ b_mu,
        const float* __restrict__ W_var, const float* __restrict__ b_var,
        const float* __restrict__ style_L,
        float* __restrict__ out_mu, float* __restrict__ out_var,
        __bf16* __restrict__ Ahi, __bf16* __restrict__ Alo,
        __bf16* __restrict__ Yhi, __bf16* __restrict__ Ylo) {
    __shared__ float yrow[Y_DIM];
    __shared__ float redM[8];
    __shared__ float redS[8];

    const int b = blockIdx.x;
    const int k = threadIdx.x;           // 0..255
    const int lane = k & 31, wave = k >> 5;

    if (k < Y_DIM) {
        float v = y[b * Y_DIM + k];
        yrow[k] = v;
        __bf16 h, l;
        split_bf16(v, h, l);
        Yhi[b * Y_DIM + k] = h;
        Ylo[b * Y_DIM + k] = l;
    }
    __syncthreads();

    float mu = b_mu[k];
    float vp = b_var[k];
    float tl = 0.0f;
#pragma unroll 8
    for (int j = 0; j < Y_DIM; ++j) {
        float yj = yrow[j];
        mu = fmaf(yj, W_mu[k * Y_DIM + j], mu);
        vp = fmaf(yj, W_var[k * Y_DIM + j], vp);
        tl = fmaf(yj, style_L[j * Z_DIM + k], tl);
    }
    out_mu[b * Z_DIM + k] = mu;
    // stable softplus
    out_var[b * Z_DIM + k] = fmaxf(vp, 0.0f) + log1pf(__expf(-fabsf(vp)));

    // softmax over z row (256 wide)
    float t = z[b * Z_DIM + k];
    float m = t;
#pragma unroll
    for (int o = 16; o > 0; o >>= 1) m = fmaxf(m, __shfl_xor(m, o, 32));
    if (lane == 0) redM[wave] = m;
    __syncthreads();
    float M = redM[0];
#pragma unroll
    for (int i = 1; i < 8; ++i) M = fmaxf(M, redM[i]);

    float e = __expf(t - M);
    float s = e;
#pragma unroll
    for (int o = 16; o > 0; o >>= 1) s += __shfl_xor(s, o, 32);
    if (lane == 0) redS[wave] = s;
    __syncthreads();
    float S = redS[0];
#pragma unroll
    for (int i = 1; i < 8; ++i) S += redS[i];

    float A = (e / S) * tl;
    __bf16 h, l;
    split_bf16(A, h, l);
    Ahi[b * Z_DIM + k] = h;
    Alo[b * Z_DIM + k] = l;
}

// ---------------------------------------------------------------------------
// Kernel 4: fused WMMA GEMM.
//   h_pre[b,x] = sum_k A[b,k]  * W_g2[x,k]     (K = 256)
//   tmp_R[b,x] = sum_j y[b,j]  * Rt[x,j]       (K = 64)
//   h = tmp_R * h_pre  -> written to hout (d_out x_rec region, used as scratch)
//   column sums / sumsq reduced through LDS, then one global atomic per column.
// Block = 256 threads = 8 waves; block tile 32(M) x 128(N).
// Each wave owns one 16-row M slice and TWO 16x16 N tiles -> the A hi/lo
// operands are loaded once and reused for both B tiles (6 WMMA per 12 loads).
// 3-term bf16 split per k-step: hi*hi + hi*lo + lo*hi  (~fp32 accuracy).
// ---------------------------------------------------------------------------
__global__ void __launch_bounds__(256)
k_gemm(const __bf16* __restrict__ Ahi, const __bf16* __restrict__ Alo,
       const __bf16* __restrict__ Whi, const __bf16* __restrict__ Wlo,
       const __bf16* __restrict__ Yhi, const __bf16* __restrict__ Ylo,
       const __bf16* __restrict__ Rhi, const __bf16* __restrict__ Rlo,
       float* __restrict__ hout, float* __restrict__ s1, float* __restrict__ s2) {
    __shared__ float sm1[128];
    __shared__ float sm2[128];

    const int tid  = threadIdx.x;
    const int lane = tid & 31;
    const int wave = tid >> 5;
    const int wm = wave & 1;        // 0..1  -> M sub-tile
    const int wq = wave >> 1;       // 0..3  -> N pair (32 columns)
    const int m0 = blockIdx.y * 32 + wm * 16;
    const int n0 = blockIdx.x * 128 + wq * 32;

    const int rN = lane & 15;       // row-in-tile for A, col-in-tile for B/C
    const int g  = lane >> 4;       // half-wave select

    const int mrow  = m0 + rN;                           // A / Y row
    const int ncol0 = n0 + rN;                           // output col, tile 0
    const int ncol1 = n0 + 16 + rN;                      // output col, tile 1
    const int nck0  = ncol0 < X_DIM ? ncol0 : X_DIM - 1;
    const int nck1  = ncol1 < X_DIM ? ncol1 : X_DIM - 1;

    if (tid < 128) { sm1[tid] = 0.0f; sm2[tid] = 0.0f; }
    __syncthreads();

    const __bf16* Ah  = Ahi + (size_t)mrow * Z_DIM;
    const __bf16* Al  = Alo + (size_t)mrow * Z_DIM;
    const __bf16* Wh0 = Whi + (size_t)nck0 * Z_DIM;
    const __bf16* Wl0 = Wlo + (size_t)nck0 * Z_DIM;
    const __bf16* Wh1 = Whi + (size_t)nck1 * Z_DIM;
    const __bf16* Wl1 = Wlo + (size_t)nck1 * Z_DIM;
    const __bf16* Yh  = Yhi + (size_t)mrow * Y_DIM;
    const __bf16* Yl  = Ylo + (size_t)mrow * Y_DIM;
    const __bf16* Rh0 = Rhi + (size_t)nck0 * Y_DIM;
    const __bf16* Rl0 = Rlo + (size_t)nck0 * Y_DIM;
    const __bf16* Rh1 = Rhi + (size_t)nck1 * Y_DIM;
    const __bf16* Rl1 = Rlo + (size_t)nck1 * Y_DIM;

    // Warm the K=64 operand lines while the K=256 loop runs (global_prefetch_b8)
    __builtin_prefetch(Yh, 0, 0);
    __builtin_prefetch(Yl, 0, 0);
    __builtin_prefetch(Rh0, 0, 0);
    __builtin_prefetch(Rh1, 0, 0);

    v8f c1a = {};   // h_pre accumulator, tile 0
    v8f c1b = {};   // h_pre accumulator, tile 1
    v8f c2a = {};   // tmp_R accumulator, tile 0
    v8f c2b = {};   // tmp_R accumulator, tile 1

#pragma unroll
    for (int kk = 0; kk < Z_DIM; kk += 32) {
        // A operand (16x32 bf16): per-lane chunks at k+g*8 and k+16+g*8
        v16bf ah = ld16(Ah + kk + g * 8, Ah + kk + 16 + g * 8);
        v16bf al = ld16(Al + kk + g * 8, Al + kk + 16 + g * 8);
        // B operands (32x16 bf16): per-lane chunks at k+g*8 and k+16+g*8
        v16bf bh0 = ld16(Wh0 + kk + g * 8, Wh0 + kk + 16 + g * 8);
        v16bf bl0 = ld16(Wl0 + kk + g * 8, Wl0 + kk + 16 + g * 8);
        v16bf bh1 = ld16(Wh1 + kk + g * 8, Wh1 + kk + 16 + g * 8);
        v16bf bl1 = ld16(Wl1 + kk + g * 8, Wl1 + kk + 16 + g * 8);
        c1a = wmma_bf16(ah, bh0, c1a);
        c1a = wmma_bf16(ah, bl0, c1a);
        c1a = wmma_bf16(al, bh0, c1a);
        c1b = wmma_bf16(ah, bh1, c1b);
        c1b = wmma_bf16(ah, bl1, c1b);
        c1b = wmma_bf16(al, bh1, c1b);
    }

#pragma unroll
    for (int kk = 0; kk < Y_DIM; kk += 32) {
        v16bf ah = ld16(Yh + kk + g * 8, Yh + kk + 16 + g * 8);
        v16bf al = ld16(Yl + kk + g * 8, Yl + kk + 16 + g * 8);
        v16bf bh0 = ld16(Rh0 + kk + g * 8, Rh0 + kk + 16 + g * 8);
        v16bf bl0 = ld16(Rl0 + kk + g * 8, Rl0 + kk + 16 + g * 8);
        v16bf bh1 = ld16(Rh1 + kk + g * 8, Rh1 + kk + 16 + g * 8);
        v16bf bl1 = ld16(Rl1 + kk + g * 8, Rl1 + kk + 16 + g * 8);
        c2a = wmma_bf16(ah, bh0, c2a);
        c2a = wmma_bf16(ah, bl0, c2a);
        c2a = wmma_bf16(al, bh0, c2a);
        c2b = wmma_bf16(ah, bh1, c2b);
        c2b = wmma_bf16(ah, bl1, c2b);
        c2b = wmma_bf16(al, bh1, c2b);
    }

    // C layout: lane L holds column N=L%16; reg r holds row m0 + r + (L/16)*8.
    float s0a = 0.0f, s0b = 0.0f, q0a = 0.0f, q0b = 0.0f;
#pragma unroll
    for (int r = 0; r < 8; ++r) {
        int row = m0 + r + g * 8;
        float ha = c2a[r] * c1a[r];
        float hb = c2b[r] * c1b[r];
        if (ncol0 < X_DIM) hout[(size_t)row * X_DIM + ncol0] = ha;
        if (ncol1 < X_DIM) hout[(size_t)row * X_DIM + ncol1] = hb;
        s0a += ha; q0a = fmaf(ha, ha, q0a);
        s0b += hb; q0b = fmaf(hb, hb, q0b);
    }
    // lanes L and L^16 share the same column -> combine across half-waves
    s0a += __shfl_xor(s0a, 16, 32);
    q0a += __shfl_xor(q0a, 16, 32);
    s0b += __shfl_xor(s0b, 16, 32);
    q0b += __shfl_xor(q0b, 16, 32);
    if (g == 0) {
        int lc0 = wq * 32 + rN;         // block-local column, tile 0
        int lc1 = wq * 32 + 16 + rN;    // block-local column, tile 1
        atomicAdd(&sm1[lc0], s0a);      // ds_add_f32
        atomicAdd(&sm2[lc0], q0a);
        atomicAdd(&sm1[lc1], s0b);
        atomicAdd(&sm2[lc1], q0b);
    }
    __syncthreads();
    if (tid < 128) {
        int x = blockIdx.x * 128 + tid;
        if (x < X_DIM) {
            atomicAdd(&s1[x], sm1[tid]);
            atomicAdd(&s2[x], sm2[tid]);
        }
    }
}

// ---------------------------------------------------------------------------
// Kernel 5: finalize batch-norm column stats into affine coeffs:
//   t = h*cg + cb  where cg = gamma*rstd, cb = beta - mean*gamma*rstd
// ---------------------------------------------------------------------------
__global__ void k_stats(const float* __restrict__ s1, const float* __restrict__ s2,
                        const float* __restrict__ gamma, const float* __restrict__ beta,
                        float* __restrict__ cg, float* __restrict__ cb) {
    int x = blockIdx.x * blockDim.x + threadIdx.x;
    if (x >= X_DIM) return;
    const float invB = 1.0f / (float)B_DIM;
    float mean = s1[x] * invB;
    float var  = s2[x] * invB - mean * mean;
    float rstd = rsqrtf(var + BN_EPS);
    float gsc  = rstd * gamma[x];
    cg[x] = gsc;
    cb[x] = beta[x] - mean * gsc;
}

// ---------------------------------------------------------------------------
// Kernel 6: per-row softmax over X (in place on d_out x_rec region).
// One 1024-thread block per row. Each thread owns ceil(X/1024)=30 elements,
// cached in registers across the max/sum phases: the 120KB row is read from
// global exactly ONCE and written exactly ONCE (true HBM minimum).
// ---------------------------------------------------------------------------
#define SM_IT ((X_DIM + 1023) / 1024)   // 30

__global__ void __launch_bounds__(1024)
k_softmax(float* __restrict__ h, const float* __restrict__ cg,
          const float* __restrict__ cb) {
    __shared__ float red[32];
    const int b = blockIdx.x;
    const int tid = threadIdx.x, lane = tid & 31, wave = tid >> 5;
    float* row = h + (size_t)b * X_DIM;

    float tv[SM_IT];

    // phase 1: load + affine, keep in registers, block max
    float m = -3.4e38f;
#pragma unroll
    for (int i = 0; i < SM_IT; ++i) {
        int x = tid + i * 1024;
        float t = -3.4e38f;
        if (x < X_DIM) t = fmaf(row[x], cg[x], cb[x]);
        tv[i] = t;
        m = fmaxf(m, t);
    }
#pragma unroll
    for (int o = 16; o > 0; o >>= 1) m = fmaxf(m, __shfl_xor(m, o, 32));
    if (lane == 0) red[wave] = m;
    __syncthreads();
    if (tid < 32) {
        float v = red[lane];
#pragma unroll
        for (int o = 16; o > 0; o >>= 1) v = fmaxf(v, __shfl_xor(v, o, 32));
        if (lane == 0) red[0] = v;
    }
    __syncthreads();
    const float M = red[0];
    __syncthreads();   // protect red before reuse

    // phase 2: exp from registers, block sum
    float s = 0.0f;
#pragma unroll
    for (int i = 0; i < SM_IT; ++i) {
        float e = __expf(tv[i] - M);   // OOB slots: exp(-inf) = 0
        tv[i] = e;
        s += e;
    }
#pragma unroll
    for (int o = 16; o > 0; o >>= 1) s += __shfl_xor(s, o, 32);
    if (lane == 0) red[wave] = s;
    __syncthreads();
    if (tid < 32) {
        float v = red[lane];
#pragma unroll
        for (int o = 16; o > 0; o >>= 1) v += __shfl_xor(v, o, 32);
        if (lane == 0) red[0] = v;
    }
    __syncthreads();
    const float inv = 1.0f / red[0];

    // phase 3: single write of the normalized row
#pragma unroll
    for (int i = 0; i < SM_IT; ++i) {
        int x = tid + i * 1024;
        if (x < X_DIM) row[x] = tv[i] * inv;
    }
}

// ---------------------------------------------------------------------------
// Launcher
// ---------------------------------------------------------------------------
extern "C" void kernel_launch(void* const* d_in, const int* in_sizes, int n_in,
                              void* d_out, int out_size, void* d_ws, size_t ws_size,
                              hipStream_t stream) {
    const float* z       = (const float*)d_in[0];
    const float* y       = (const float*)d_in[1];
    const float* W_mu    = (const float*)d_in[2];
    const float* b_mu    = (const float*)d_in[3];
    const float* W_var   = (const float*)d_in[4];
    const float* b_var   = (const float*)d_in[5];
    const float* W_g2    = (const float*)d_in[6];
    const float* style_L = (const float*)d_in[7];
    const float* style_R = (const float*)d_in[8];
    const float* gamma   = (const float*)d_in[9];
    const float* beta    = (const float*)d_in[10];

    float* out    = (float*)d_out;
    float* out_mu = out;                                // [B,Z]
    float* out_vr = out + (size_t)B_DIM * Z_DIM;        // [B,Z]
    float* out_x  = out + 2 * (size_t)B_DIM * Z_DIM;    // [B,X] (also h scratch)

    // ---- workspace carve-up (all 256B aligned) ----
    char* p = (char*)d_ws;
    auto carve = [&](size_t bytes) {
        char* r = p;
        p += (bytes + 255) & ~(size_t)255;
        return r;
    };
    __bf16* Whi = (__bf16*)carve((size_t)X_DIM * Z_DIM * 2);
    __bf16* Wlo = (__bf16*)carve((size_t)X_DIM * Z_DIM * 2);
    __bf16* Rhi = (__bf16*)carve((size_t)X_DIM * Y_DIM * 2);
    __bf16* Rlo = (__bf16*)carve((size_t)X_DIM * Y_DIM * 2);
    __bf16* Ahi = (__bf16*)carve((size_t)B_DIM * Z_DIM * 2);
    __bf16* Alo = (__bf16*)carve((size_t)B_DIM * Z_DIM * 2);
    __bf16* Yhi = (__bf16*)carve((size_t)B_DIM * Y_DIM * 2);
    __bf16* Ylo = (__bf16*)carve((size_t)B_DIM * Y_DIM * 2);
    float*  s1  = (float*)carve((size_t)X_DIM * 4);
    float*  s2  = (float*)carve((size_t)X_DIM * 4);
    float*  cg  = (float*)carve((size_t)X_DIM * 4);
    float*  cb  = (float*)carve((size_t)X_DIM * 4);
    (void)ws_size; (void)in_sizes; (void)n_in; (void)out_size;

    // 1) weight conversions
    {
        int n = X_DIM * Z_DIM;
        k_conv_w<<<(n + 255) / 256, 256, 0, stream>>>(W_g2, Whi, Wlo, n);
    }
    // 2) style_R transpose + stat zeroing
    k_prep_R<<<(X_DIM + 255) / 256, 256, 0, stream>>>(style_R, Rhi, Rlo, s1, s2);
    // 3) front end (y_mu, y_var, softmax z, A, y->bf16)
    k_front<<<B_DIM, Z_DIM, 0, stream>>>(z, y, W_mu, b_mu, W_var, b_var, style_L,
                                         out_mu, out_vr, Ahi, Alo, Yhi, Ylo);
    // 4) fused WMMA GEMM -> h + column stats
    {
        dim3 grid((X_DIM + 127) / 128, B_DIM / 32);
        k_gemm<<<grid, 256, 0, stream>>>(Ahi, Alo, Whi, Wlo, Yhi, Ylo, Rhi, Rlo,
                                         out_x, s1, s2);
    }
    // 5) batch-norm coefficients
    k_stats<<<(X_DIM + 255) / 256, 256, 0, stream>>>(s1, s2, gamma, beta, cg, cb);
    // 6) per-row softmax in place
    k_softmax<<<B_DIM, 1024, 0, stream>>>(out_x, cg, cb);
}
